// SNN_AC_83451214561486
// MI455X (gfx1250) — matrix operand order repaired and checked by reading, not compiled, not run on hardware
//
#include <hip/hip_runtime.h>
#include <hip/hip_bf16.h>

typedef __attribute__((ext_vector_type(16))) _Float16 v16h;
typedef __attribute__((ext_vector_type(8)))  _Float16 v8h;
typedef __attribute__((ext_vector_type(8)))  float    v8f;

// ---------------------------------------------------------------------------
// f32 -> f16 (spikes {0,1} are exact in f16; weights ~5e-4 rel err, f32 accum)
// ---------------------------------------------------------------------------
__global__ void f32_to_f16_kernel(const float* __restrict__ in,
                                  _Float16* __restrict__ out, int n) {
  int i = blockIdx.x * blockDim.x + threadIdx.x;
  if (i < n) out[i] = (_Float16)in[i];
}

// ---------------------------------------------------------------------------
// WMMA GEMM with LDS-staged B: C[M,N] = A[M,K] * Bt[N,K]^T + bias[N]
// Workgroup: 256 thr = 8 wave32, tile 128(M) x 128(N).
// Wave (w&3) -> 32-row strip (2 M-tiles), (w>>2) -> 64-col strip (4 N-tiles)
// => 8 v_wmma_f32_16x16x32_f16 per wave per K-step, A reused x4, B reused x2.
// B tile (128 x 32 f16) double-buffered in LDS, loaded once per WG per K-step.
// LDS row stride = 40 halfs (80B): lane bank = (20*l16)%64 -> conflict-free.
// Requires M%128==0, N%128==0, K%32==0 (true for all calls here).
// ---------------------------------------------------------------------------
#define BM 128
#define BN 128
#define BK 32
#define BPAD 8

__global__ __launch_bounds__(256) void wmma_gemm_kernel(
    const _Float16* __restrict__ A,
    const _Float16* __restrict__ Bt,
    const float* __restrict__ bias,
    float* __restrict__ C,
    int M, int N, int K) {
  __shared__ _Float16 Bs[2][BN][BK + BPAD];

  const int tid  = threadIdx.x;
  const int lane = tid & 31;
  const int wave = tid >> 5;
  const int l16  = lane & 15;
  const int half = lane >> 4;          // 0: lanes 0-15, 1: lanes 16-31
  const int m_w  = wave & 3;           // 32-row strip within block tile
  const int n_w  = wave >> 2;          // 64-col strip within block tile

  const int blockM = blockIdx.y * BM;
  const int blockN = blockIdx.x * BN;

  v8f acc[2][4] = {};

  // A fragment addressing (ISA 7.12.2 16-bit A 16x32):
  // lane<16: row=l16, K 0..7 & 16..23 ; lane>=16: row=l16, K 8..15 & 24..31
  const _Float16* Arow0 = A + (size_t)(blockM + m_w * 32 +  0 + l16) * K + half * 8;
  const _Float16* Arow1 = A + (size_t)(blockM + m_w * 32 + 16 + l16) * K + half * 8;

  // B staging: 256 threads x 32B = 8KB = 128 rows x 32 halfs.
  const int brow = tid >> 1;                 // 0..127
  const int boff = (tid & 1) * 16;           // halfs
  const _Float16* Bsrc = Bt + (size_t)(blockN + brow) * K + boff;

  { // preload K-chunk 0
    v8h t0 = *(const v8h*)(Bsrc + 0);
    v8h t1 = *(const v8h*)(Bsrc + 8);
    *(v8h*)(&Bs[0][brow][boff + 0]) = t0;
    *(v8h*)(&Bs[0][brow][boff + 8]) = t1;
  }
  __syncthreads();

  int cur = 0;
  for (int kk = 0; kk < K; kk += BK) {
    const bool has_next = (kk + BK) < K;
    v8h p0{}, p1{};
    if (has_next) {                          // prefetch next B chunk to regs
      p0 = *(const v8h*)(Bsrc + kk + BK + 0);
      p1 = *(const v8h*)(Bsrc + kk + BK + 8);
    }

    // A fragments for this wave's two M-tiles (direct global, L2-resident)
    v16h a0, a1;
    {
      v8h lo0 = *(const v8h*)(Arow0 + kk);
      v8h hi0 = *(const v8h*)(Arow0 + kk + 16);
      v8h lo1 = *(const v8h*)(Arow1 + kk);
      v8h hi1 = *(const v8h*)(Arow1 + kk + 16);
#pragma unroll
      for (int i = 0; i < 8; ++i) {
        a0[i] = lo0[i]; a0[8 + i] = hi0[i];
        a1[i] = lo1[i]; a1[8 + i] = hi1[i];
      }
    }

#pragma unroll
    for (int nt = 0; nt < 4; ++nt) {
      // B fragment from LDS: lane n holds K 8h..8h+7 & 16+8h..16+8h+7 of column n
      const _Float16* bp = &Bs[cur][n_w * 64 + nt * 16 + l16][half * 8];
      v8h blo = *(const v8h*)(bp);
      v8h bhi = *(const v8h*)(bp + 16);
      v16h b;
#pragma unroll
      for (int i = 0; i < 8; ++i) { b[i] = blo[i]; b[8 + i] = bhi[i]; }

      acc[0][nt] = __builtin_amdgcn_wmma_f32_16x16x32_f16(
          false, a0, false, b, (short)0, acc[0][nt], false, false);
      acc[1][nt] = __builtin_amdgcn_wmma_f32_16x16x32_f16(
          false, a1, false, b, (short)0, acc[1][nt], false, false);
    }

    if (has_next) {                          // fill the other buffer
      *(v8h*)(&Bs[cur ^ 1][brow][boff + 0]) = p0;
      *(v8h*)(&Bs[cur ^ 1][brow][boff + 8]) = p1;
    }
    __syncthreads();
    cur ^= 1;
  }

  // C layout (ISA 7.12.2): VGPR r -> row = tile + r + 8*half, col = tile + l16
#pragma unroll
  for (int mt = 0; mt < 2; ++mt) {
    const int orow = blockM + m_w * 32 + mt * 16 + half * 8;
#pragma unroll
    for (int nt = 0; nt < 4; ++nt) {
      const int col = blockN + n_w * 64 + nt * 16 + l16;
      const float bv = bias[col];
#pragma unroll
      for (int r = 0; r < 8; ++r) {
        C[(size_t)(orow + r) * N + col] = acc[mt][nt][r] + bv;
      }
    }
  }
}

// ---------------------------------------------------------------------------
// m1 recurrence for k in [kb,ke): independent of m2, so spikes for a whole
// chunk are produced in ONE launch (thread-local k loop).
//   reset = (m_prev > 1); m = beta*m_prev + cur - reset; spk = (m - 1 > 0)
// ---------------------------------------------------------------------------
__global__ void spike1_scan_kernel(const float* __restrict__ cur1,
                                   float* __restrict__ m1s,
                                   const float* __restrict__ minit,
                                   const float* __restrict__ beta_p,
                                   _Float16* __restrict__ s1_all,
                                   int n, int kb, int ke) {
  int i = blockIdx.x * blockDim.x + threadIdx.x;
  if (i >= n) return;
  const float beta = *beta_p;
  const float c = cur1[i];
  float m = (kb == 0) ? minit[i] : m1s[i];
  for (int k = kb; k < ke; ++k) {
    const float reset = (m > 1.0f) ? 1.0f : 0.0f;
    m = beta * m + c - reset;
    s1_all[(size_t)(k - kb) * n + i] =
        (m - 1.0f) > 0.0f ? (_Float16)1.0f : (_Float16)0.0f;
  }
  m1s[i] = m;
}

// m2 recurrence over stacked h chunk; emits s2 (final chunk's write survives)
__global__ void m2_scan_kernel(const float* __restrict__ h_all,
                               float* __restrict__ m2s,
                               const float* __restrict__ minit,
                               const float* __restrict__ beta_p,
                               _Float16* __restrict__ s2h,
                               int n, int kb, int ke) {
  int i = blockIdx.x * blockDim.x + threadIdx.x;
  if (i >= n) return;
  const float beta = *beta_p;
  float m = (kb == 0) ? minit[i] : m2s[i];
  for (int k = kb; k < ke; ++k) {
    const float reset = (m > 1.0f) ? 1.0f : 0.0f;
    m = beta * m + h_all[(size_t)(k - kb) * n + i] - reset;
  }
  m2s[i] = m;
  s2h[i] = (m - 1.0f) > 0.0f ? (_Float16)1.0f : (_Float16)0.0f;
}

// ---------------------------------------------------------------------------
// Head: one wave32 per output. logits = X[row,:].W[j,:] + bias[j], then Leaky.
// ---------------------------------------------------------------------------
__global__ __launch_bounds__(256) void head_kernel(
    const _Float16* __restrict__ X, const float* __restrict__ W,
    const float* __restrict__ bias, const float* __restrict__ mem_in,
    const float* __restrict__ beta_p, float* __restrict__ out_spk,
    float* __restrict__ out_mem, int M, int Nout, int K) {
  const int gw   = (int)((blockIdx.x * blockDim.x + threadIdx.x) >> 5);
  const int lane = threadIdx.x & 31;
  if (gw >= M * Nout) return;             // uniform per wave
  const int row = gw / Nout;
  const int j   = gw % Nout;

  const _Float16* xr = X + (size_t)row * K;
  const float*    wr = W + (size_t)j * K;
  float s = 0.0f;
  for (int k = lane; k < K; k += 32) s += (float)xr[k] * wr[k];
#pragma unroll
  for (int off = 16; off; off >>= 1) s += __shfl_xor(s, off, 32);

  if (lane == 0) {
    const float beta  = *beta_p;
    const float mp    = mem_in[(size_t)row * Nout + j];
    const float reset = (mp > 1.0f) ? 1.0f : 0.0f;
    const float mn    = beta * mp + (s + bias[j]) - reset;
    out_mem[(size_t)row * Nout + j] = mn;
    out_spk[(size_t)row * Nout + j] = (mn - 1.0f) > 0.0f ? 1.0f : 0.0f;
  }
}

// ---------------------------------------------------------------------------
extern "C" void kernel_launch(void* const* d_in, const int* in_sizes, int n_in,
                              void* d_out, int out_size, void* d_ws, size_t ws_size,
                              hipStream_t stream) {
  const float* inputs   = (const float*)d_in[0];
  const float* mem1     = (const float*)d_in[1];
  const float* mem2     = (const float*)d_in[2];
  const float* mem_act  = (const float*)d_in[3];
  const float* mem_crit = (const float*)d_in[4];
  const float* W1 = (const float*)d_in[5];
  const float* b1 = (const float*)d_in[6];
  const float* W2 = (const float*)d_in[7];
  const float* b2 = (const float*)d_in[8];
  const float* Wc = (const float*)d_in[9];
  const float* bc = (const float*)d_in[10];
  const float* Wa = (const float*)d_in[11];
  const float* ba = (const float*)d_in[12];
  const float* beta1     = (const float*)d_in[13];
  const float* beta2     = (const float*)d_in[14];
  const float* beta_crit = (const float*)d_in[15];
  const float* beta_act  = (const float*)d_in[16];

  constexpr int T = 16, S = 512, H = 2048, A = 32;
  constexpr int M  = 16 * 64;            // T*B = 1024 rows per step
  constexpr int MH = M * H;

  // ---- output layout (flat concat, reference return order) ----
  float* out        = (float*)d_out;
  float* out_value  = out;
  float* out_action = out_value + (size_t)M;
  float* out_m1     = out_action + (size_t)M * A;
  float* out_m2     = out_m1 + (size_t)MH;
  float* out_mact   = out_m2 + (size_t)MH;
  float* out_mcrit  = out_mact + (size_t)M * A;

  // ---- workspace: fixed part + per-step chunk part ----
  const size_t fixedB =
      (size_t)M * S * 2 + (size_t)H * S * 2 + (size_t)H * H * 2 +  // inh, W1h, W2h
      3 * (size_t)MH * 4 +                                         // cur1, m1s, m2s
      (size_t)MH * 2;                                              // s2h
  const size_t perk = (size_t)MH * (2 + 4);                        // s1 f16 + h f32
  int C = 1;                                                       // fusion chunk
  if (ws_size > fixedB + perk) {
    size_t c = (ws_size - fixedB) / perk;
    C = (c >= (size_t)T) ? T : (int)c;
  }

  char* w = (char*)d_ws;
  _Float16* inh    = (_Float16*)w; w += (size_t)M * S * 2;
  _Float16* W1h    = (_Float16*)w; w += (size_t)H * S * 2;
  _Float16* W2h    = (_Float16*)w; w += (size_t)H * H * 2;
  float*    cur1   = (float*)w;    w += (size_t)MH * 4;
  float*    m1s    = (float*)w;    w += (size_t)MH * 4;
  float*    m2s    = (float*)w;    w += (size_t)MH * 4;
  _Float16* s2h    = (_Float16*)w; w += (size_t)MH * 2;
  _Float16* s1_all = (_Float16*)w; w += (size_t)C * MH * 2;
  float*    h_all  = (float*)w;    w += (size_t)C * MH * 4;

  const dim3 blk(256);

  f32_to_f16_kernel<<<dim3((M * S + 255) / 256), blk, 0, stream>>>(inputs, inh, M * S);
  f32_to_f16_kernel<<<dim3((H * S + 255) / 256), blk, 0, stream>>>(W1, W1h, H * S);
  f32_to_f16_kernel<<<dim3((H * H + 255) / 256), blk, 0, stream>>>(W2, W2h, H * H);

  // cur1 = inputs @ W1^T + b1   (M=1024, N=2048, K=512)
  wmma_gemm_kernel<<<dim3(H / BN, M / BM), blk, 0, stream>>>(inh, W1h, b1, cur1, M, H, S);

  // Fused scan: chunks of C steps -> one big GEMM of M = C*1024 per chunk
  const dim3 gelem((MH + 255) / 256);
  for (int kb = 0; kb < T;) {
    const int ke = (kb + C < T) ? (kb + C) : T;
    const int c  = ke - kb;
    spike1_scan_kernel<<<gelem, blk, 0, stream>>>(cur1, m1s, mem1, beta1, s1_all, MH, kb, ke);
    wmma_gemm_kernel<<<dim3(H / BN, (c * M) / BM), blk, 0, stream>>>(
        s1_all, W2h, b2, h_all, c * M, H, H);
    m2_scan_kernel<<<gelem, blk, 0, stream>>>(h_all, m2s, mem2, beta2, s2h, MH, kb, ke);
    kb = ke;
  }

  // heads on final s2
  head_kernel<<<dim3((M * 1 * 32 + 255) / 256), blk, 0, stream>>>(
      s2h, Wc, bc, mem_crit, beta_crit, out_value, out_mcrit, M, 1, H);
  head_kernel<<<dim3((M * A * 32 + 255) / 256), blk, 0, stream>>>(
      s2h, Wa, ba, mem_act, beta_act, out_action, out_mact, M, A, H);

  // final membrane states -> outputs (async d2d, graph-capture safe)
  hipMemcpyAsync(out_m1, m1s, (size_t)MH * 4, hipMemcpyDeviceToDevice, stream);
  hipMemcpyAsync(out_m2, m2s, (size_t)MH * 4, hipMemcpyDeviceToDevice, stream);
}
